// Net_57870389346502
// MI455X (gfx1250) — matrix operand order repaired and checked by reading, not compiled, not run on hardware
//
#include <hip/hip_runtime.h>

// ---------------------------------------------------------------------------
// R-GCN link predictor on MI455X (gfx1250, wave32, WMMA).
// - All dense math via v_wmma_f32_16x16x32_bf16 (bf16 operands, f32 accum).
// - Per-edge matvecs restructured into per-node GEMMs (y_r = x @ W_r), so the
//   edge phase is pure gather + invcnt-scaled f32 atomic scatter (L2-resident).
// - y_r stored in bf16: halves the dominant random-gather stream (64B/edge).
// - Relation in-degree counts are layer-invariant: computed once, inverted,
//   folded into the scatter.
// ---------------------------------------------------------------------------

#define N_PAPER 60000
#define N_MESH  30000
#define N_NODES 90000
#define E_TRAIN 2000000
#define E_DEC   400000
#define NTILES  (N_NODES / 16)   // 5625 row tiles of 16

typedef __attribute__((ext_vector_type(16))) __bf16 v16bf;
typedef __attribute__((ext_vector_type(8)))  float  v8f;

// --------------------------- small utility kernels -------------------------

__global__ void zero_kernel(float* __restrict__ p, int n) {
    int i = blockIdx.x * blockDim.x + threadIdx.x;
    if (i < n) p[i] = 0.0f;
}

// cnt[r][n] = number of incoming edges of relation r at node n (layer-invariant)
__global__ void count_kernel(const int* __restrict__ ei, const int* __restrict__ et,
                             float* __restrict__ cnt) {
    int e = blockIdx.x * blockDim.x + threadIdx.x;
    if (e >= E_TRAIN) return;
    int d = ei[E_TRAIN + e];
    int r = et[e];
    atomicAdd(&cnt[r * N_NODES + d], 1.0f);
}

__global__ void invert_kernel(const float* __restrict__ cnt, float* __restrict__ inv, int n) {
    int i = blockIdx.x * blockDim.x + threadIdx.x;
    if (i < n) inv[i] = 1.0f / fmaxf(cnt[i], 1.0f);
}

// --------------------------- weight packing --------------------------------
// wpack layout: [mat][k(0..31)][n(0..31)] bf16, zero padded; mats 0,1 = W_r,
// mat 2 = root. bpack: [mat][n(0..31)] f32 bias.

__global__ void wprep_rgcn_kernel(const float* __restrict__ bases,  // [NB, di, do]
                                  const float* __restrict__ comp,   // [NREL, NB]
                                  const float* __restrict__ root,   // [di, do]
                                  const float* __restrict__ bias,   // [do]
                                  int di, int doo,
                                  __bf16* __restrict__ wpack, float* __restrict__ bpack) {
    int tid = blockIdx.x * blockDim.x + threadIdx.x;
    if (tid < 3 * 32 * 32) {
        int mat = tid >> 10, k = (tid >> 5) & 31, n = tid & 31;
        float v = 0.0f;
        if (k < di && n < doo) {
            if (mat < 2) {
                #pragma unroll
                for (int b = 0; b < 4; ++b)
                    v += comp[mat * 4 + b] * bases[(b * di + k) * doo + n];
            } else {
                v = root[k * doo + n];
            }
        }
        wpack[tid] = (__bf16)v;
    }
    if (tid < 3 * 32) {
        int mat = tid >> 5, n = tid & 31;
        float v = 0.0f;
        if (mat == 2 && n < doo) v = bias[n];
        bpack[tid] = v;
    }
}

// Decoder weights: mats 0,1 = Wd0, Wd1 (16x16) with their biases.
__global__ void wprep_decode_kernel(const float* __restrict__ Wd0, const float* __restrict__ bd0,
                                    const float* __restrict__ Wd1, const float* __restrict__ bd1,
                                    __bf16* __restrict__ wpack, float* __restrict__ bpack) {
    int tid = blockIdx.x * blockDim.x + threadIdx.x;
    if (tid < 3 * 32 * 32) {
        int mat = tid >> 10, k = (tid >> 5) & 31, n = tid & 31;
        float v = 0.0f;
        if (mat < 2 && k < 16 && n < 16) v = (mat == 0 ? Wd0 : Wd1)[k * 16 + n];
        wpack[tid] = (__bf16)v;
    }
    if (tid < 3 * 32) {
        int mat = tid >> 5, n = tid & 31;
        float v = 0.0f;
        if (mat == 0 && n < 16) v = bd0[n];
        if (mat == 1 && n < 16) v = bd1[n];
        bpack[tid] = v;
    }
}

// --------------------------- WMMA fragment helpers -------------------------
// A fragment (16x32 bf16):  lane L -> row M = L&15; element e -> K = (e<8?e:e+8)+8*(L>>4)
// B fragment (32x16 bf16):  lane L -> col N = L&15; element e -> K = (e<8?e:e+8)+8*(L>>4)
// D fragment (16x16 f32):   lane L, element e -> row M = e + 8*(L>>4), col N = L&15

// --------------------------- WMMA GEMM kernels -----------------------------
// 8 waves per block, each wave an independent 16-row tile (EXEC all-ones per
// wave; the tile guard is wave-uniform).

// Input transform: x = concat(x_paper@Wp+bp, x_mesh@Wm+bm), K=128 (4 WMMA steps).
__global__ void __launch_bounds__(256)
input_transform_kernel(const float* __restrict__ xp, const float* __restrict__ xm,
                       const float* __restrict__ Wp, const float* __restrict__ bp,
                       const float* __restrict__ Wm, const float* __restrict__ bm,
                       float* __restrict__ xout) {
    const int NT_P = N_PAPER / 16;
    int t = blockIdx.x * 8 + (threadIdx.x >> 5);
    if (t >= NTILES) return;                     // wave-uniform
    const float *X, *W, *bias;
    int row0, orow0;
    if (t < NT_P) { X = xp; W = Wp; bias = bp; row0 = t * 16;          orow0 = t * 16; }
    else          { X = xm; W = Wm; bias = bm; row0 = (t - NT_P) * 16; orow0 = N_PAPER + (t - NT_P) * 16; }

    const int L = threadIdx.x & 31, half = L >> 4, m = L & 15;
    v8f acc = {};
    for (int k0 = 0; k0 < 128; k0 += 32) {
        v16bf a, b;
        #pragma unroll
        for (int e = 0; e < 16; ++e) {
            int ka = (e < 8 ? e : e + 8) + 8 * half;
            a[e] = (__bf16)X[(size_t)(row0 + m) * 128 + k0 + ka];
            int kb = (e < 8 ? e : e + 8) + 8 * half;
            b[e] = (__bf16)W[(k0 + kb) * 16 + m];
        }
        acc = __builtin_amdgcn_wmma_f32_16x16x32_bf16(false, a, false, b, (short)0, acc,
                                                      false, false);
    }
    #pragma unroll
    for (int e = 0; e < 8; ++e) {
        int M = e + 8 * half;
        xout[(size_t)(orow0 + M) * 16 + m] = acc[e] + bias[m];
    }
}

// RGCN per-node transform: y0 = x@W0 (bf16), y1 = x@W1 (bf16),
// out = x@root + bias (f32). ReLU of the previous layer fused into A load.
__global__ void __launch_bounds__(256)
rgcn_transform_kernel(const float* __restrict__ X, int di, int doo, int relu_in,
                      const __bf16* __restrict__ wpack, const float* __restrict__ bpack,
                      __bf16* __restrict__ y0, __bf16* __restrict__ y1,
                      float* __restrict__ out) {
    int tile = blockIdx.x * 8 + (threadIdx.x >> 5);
    if (tile >= NTILES) return;                  // wave-uniform
    const int L = threadIdx.x & 31, half = L >> 4, m = L & 15;
    const int row0 = tile * 16;

    v16bf a;
    #pragma unroll
    for (int e = 0; e < 16; ++e) {
        int ka = (e < 8 ? e : e + 8) + 8 * half;
        float v = (ka < di) ? X[(size_t)(row0 + m) * di + ka] : 0.0f;
        if (relu_in) v = fmaxf(v, 0.0f);
        a[e] = (__bf16)v;
    }

    const int ntile = doo >> 4;
    for (int mat = 0; mat < 3; ++mat) {
        for (int nt = 0; nt < ntile; ++nt) {
            v16bf b;
            #pragma unroll
            for (int e = 0; e < 16; ++e) {
                int kb = (e < 8 ? e : e + 8) + 8 * half;
                b[e] = wpack[mat * 1024 + kb * 32 + nt * 16 + m];
            }
            v8f acc = {};
            acc = __builtin_amdgcn_wmma_f32_16x16x32_bf16(false, a, false, b, (short)0, acc,
                                                          false, false);
            int col = nt * 16 + m;
            if (mat < 2) {
                __bf16* yp = (mat == 0) ? y0 : y1;
                #pragma unroll
                for (int e = 0; e < 8; ++e) {
                    int M = e + 8 * half;
                    yp[(size_t)(row0 + M) * doo + col] = (__bf16)acc[e];
                }
            } else {
                float bias = bpack[2 * 32 + col];
                #pragma unroll
                for (int e = 0; e < 8; ++e) {
                    int M = e + 8 * half;
                    out[(size_t)(row0 + M) * doo + col] = acc[e] + bias;
                }
            }
        }
    }
}

// Decoder per-node transform: u_r = z @ Wd_r + bd_r (f32 out, K=16 zero-padded).
__global__ void __launch_bounds__(256)
decode_transform_kernel(const float* __restrict__ Z,
                        const __bf16* __restrict__ wpack, const float* __restrict__ bpack,
                        float* __restrict__ u0, float* __restrict__ u1) {
    int tile = blockIdx.x * 8 + (threadIdx.x >> 5);
    if (tile >= NTILES) return;                  // wave-uniform
    const int L = threadIdx.x & 31, half = L >> 4, m = L & 15;
    const int row0 = tile * 16;

    v16bf a;
    #pragma unroll
    for (int e = 0; e < 16; ++e) {
        int ka = (e < 8 ? e : e + 8) + 8 * half;
        a[e] = (__bf16)((ka < 16) ? Z[(size_t)(row0 + m) * 16 + ka] : 0.0f);
    }

    for (int mat = 0; mat < 2; ++mat) {
        v16bf b;
        #pragma unroll
        for (int e = 0; e < 16; ++e) {
            int kb = (e < 8 ? e : e + 8) + 8 * half;
            b[e] = wpack[mat * 1024 + kb * 32 + m];
        }
        v8f acc = {};
        acc = __builtin_amdgcn_wmma_f32_16x16x32_bf16(false, a, false, b, (short)0, acc,
                                                      false, false);
        float bias = bpack[mat * 32 + m];
        float* up = (mat == 0) ? u0 : u1;
        #pragma unroll
        for (int e = 0; e < 8; ++e) {
            int M = e + 8 * half;
            up[(size_t)(row0 + M) * 16 + m] = acc[e] + bias;
        }
    }
}

// --------------------------- edge kernels ----------------------------------
// out[dst] += y_{et}[src] * invcnt[et][dst]   (one lane per (edge, column);
// a wave covers one edge for do=32 -> 64B coalesced bf16 gather + 128B atomics)
__global__ void scatter_kernel(const int* __restrict__ ei, const int* __restrict__ et,
                               const __bf16* __restrict__ y0, const __bf16* __restrict__ y1,
                               const float* __restrict__ invcnt,
                               float* __restrict__ out, int doo, int shift) {
    unsigned tid = blockIdx.x * blockDim.x + threadIdx.x;
    unsigned e = tid >> shift;
    if (e >= E_TRAIN) return;
    int j = (int)(tid & ((1u << shift) - 1u));
    int s = ei[e];
    int d = ei[E_TRAIN + e];
    int r = et[e];
    const __bf16* y = (r == 0) ? y0 : y1;
    float v = (float)y[(size_t)s * doo + j] * invcnt[r * N_NODES + d];
    atomicAdd(&out[(size_t)d * doo + j], v);
}

// out[e] = dot(u_{et}[src], z[dst])   (u already includes Wd + bd)
__global__ void decode_edge_kernel(const int* __restrict__ ei, const int* __restrict__ et,
                                   const float* __restrict__ u0, const float* __restrict__ u1,
                                   const float* __restrict__ z, float* __restrict__ out) {
    int e = blockIdx.x * blockDim.x + threadIdx.x;
    if (e >= E_DEC) return;
    int s = ei[e], d = ei[E_DEC + e], r = et[e];
    const float4* u  = (const float4*)((r == 0 ? u0 : u1) + (size_t)s * 16);
    const float4* zd = (const float4*)(z + (size_t)d * 16);
    float acc = 0.0f;
    #pragma unroll
    for (int q = 0; q < 4; ++q) {
        float4 a = u[q], b = zd[q];
        acc += a.x * b.x + a.y * b.y + a.z * b.z + a.w * b.w;
    }
    out[e] = acc;
}

// --------------------------- launch ----------------------------------------

extern "C" void kernel_launch(void* const* d_in, const int* in_sizes, int n_in,
                              void* d_out, int out_size, void* d_ws, size_t ws_size,
                              hipStream_t stream) {
    (void)in_sizes; (void)n_in; (void)out_size; (void)ws_size;

    const float* xp  = (const float*)d_in[0];
    const float* xm  = (const float*)d_in[1];
    const int*   tei = (const int*)d_in[2];
    const int*   tet = (const int*)d_in[3];
    const int*   pei = (const int*)d_in[4];
    const int*   pet = (const int*)d_in[5];
    const int*   nei = (const int*)d_in[6];
    const int*   net_= (const int*)d_in[7];
    const float* Wp  = (const float*)d_in[8];
    const float* bp  = (const float*)d_in[9];
    const float* Wm  = (const float*)d_in[10];
    const float* bm  = (const float*)d_in[11];
    const float* Wd0 = (const float*)d_in[12];
    const float* bd0 = (const float*)d_in[13];
    const float* Wd1 = (const float*)d_in[14];
    const float* bd1 = (const float*)d_in[15];

    // workspace layout (all offsets 256B aligned)
    float* ws = (float*)d_ws;
    size_t off = 0;
    auto alloc = [&](size_t nfloats) {
        float* p = ws + off;
        off += (nfloats + 63) & ~(size_t)63;
        return p;
    };
    float*  bufA  = alloc((size_t)N_NODES * 32);            // activations ping (f32)
    float*  bufB  = alloc((size_t)N_NODES * 32);            // activations pong (f32)
    __bf16* y0    = (__bf16*)alloc((size_t)N_NODES * 16);   // x @ W_0  (bf16, <=32 cols)
    __bf16* y1    = (__bf16*)alloc((size_t)N_NODES * 16);   // x @ W_1  (bf16)
    float*  cnt   = alloc(2 * N_NODES);
    float*  invc  = alloc(2 * N_NODES);
    float*  u0    = alloc((size_t)N_NODES * 16);            // z @ Wd0 + bd0 (f32)
    float*  u1    = alloc((size_t)N_NODES * 16);            // z @ Wd1 + bd1 (f32)
    float*  bpack = alloc(3 * 32);
    __bf16* wpack = (__bf16*)alloc(3 * 32 * 32 / 2 + 64);

    const int TB = (NTILES + 7) / 8;   // blocks for 8-wave WMMA kernels

    // relation in-degree counts (layer-invariant) -> invcnt
    zero_kernel  <<<(2 * N_NODES + 255) / 256, 256, 0, stream>>>(cnt, 2 * N_NODES);
    count_kernel <<<(E_TRAIN + 255) / 256,     256, 0, stream>>>(tei, tet, cnt);
    invert_kernel<<<(2 * N_NODES + 255) / 256, 256, 0, stream>>>(cnt, invc, 2 * N_NODES);

    // x0 = concat(paper@Wp+bp, mesh@Wm+bm)  -> bufA [N,16]
    input_transform_kernel<<<TB, 256, 0, stream>>>(xp, xm, Wp, bp, Wm, bm, bufA);

    const int dis[4]  = {16, 32, 32, 32};
    const int dos_[4] = {32, 32, 32, 16};
    float* cur = bufA;
    float* nxt = bufB;
    for (int l = 0; l < 4; ++l) {
        const float* bases = (const float*)d_in[16 + 4 * l];
        const float* comp  = (const float*)d_in[17 + 4 * l];
        const float* root  = (const float*)d_in[18 + 4 * l];
        const float* bias  = (const float*)d_in[19 + 4 * l];

        wprep_rgcn_kernel<<<12, 256, 0, stream>>>(bases, comp, root, bias,
                                                  dis[l], dos_[l], wpack, bpack);
        // y0 = x@W0, y1 = x@W1 (bf16), nxt = x@root + bias (f32); relu fused (l>0)
        rgcn_transform_kernel<<<TB, 256, 0, stream>>>(
            cur, dis[l], dos_[l], (l > 0) ? 1 : 0, wpack, bpack, y0, y1, nxt);

        int shift = (dos_[l] == 32) ? 5 : 4;
        unsigned total = (unsigned)E_TRAIN << shift;
        scatter_kernel<<<(total + 255) / 256, 256, 0, stream>>>(
            tei, tet, y0, y1, invc, nxt, dos_[l], shift);

        float* t = cur; cur = nxt; nxt = t;
    }
    // cur == z [N,16] (no relu, matching reference)

    // decoder precompute: u_r = z @ Wd_r + bd_r
    wprep_decode_kernel<<<12, 256, 0, stream>>>(Wd0, bd0, Wd1, bd1, wpack, bpack);
    decode_transform_kernel<<<TB, 256, 0, stream>>>(cur, wpack, bpack, u0, u1);

    float* out = (float*)d_out;
    decode_edge_kernel<<<(E_DEC + 255) / 256, 256, 0, stream>>>(pei, pet, u0, u1, cur, out);
    decode_edge_kernel<<<(E_DEC + 255) / 256, 256, 0, stream>>>(nei, net_, u0, u1, cur, out + E_DEC);
}
